// GPT2Model_7241314861541
// MI455X (gfx1250) — matrix operand order repaired and checked by reading, not compiled, not run on hardware
//
#include <hip/hip_runtime.h>
#include <hip/hip_bf16.h>
#include <math.h>

// GPT-2 small, fixed sizes from the reference
#define L_  12
#define D_  768
#define H_  12
#define V_  50257
#define S_  1024
#define B_  2
#define FF_ (4 * D_)        // 3072
#define HD_ (D_ / H_)       // 64
#define M_  (B_ * S_)       // 2048 activation rows
#define NEGINF (-1e9f)

typedef __attribute__((ext_vector_type(16))) _Float16 v16h;
typedef __attribute__((ext_vector_type(8)))  _Float16 v8h;
typedef __attribute__((ext_vector_type(8)))  float    v8f;

// ---------------------------------------------------------------------------
// WMMA fragment layouts as used here (wave32; A/B k = 16*hi + h, not the ISA table order)
//
// A (16x32 f16): lane row M = lane&15.
//   half h -> K = k0 + h + (laneHi?16:0)
//   => two contiguous 8-half (16B) chunks: [k0+16*hi, +8) and [k0+16*hi+8, +8)
// B (32x16 f16): lane col N = lane&15.
//   half h -> K = k0 + h + (laneHi?16:0)  => one contiguous 32B v16h load
// C/D (16x16 f32): vgpr r -> M = r + (laneHi?8:0), N = lane&15
// ---------------------------------------------------------------------------

__device__ __forceinline__ v16h load_a_frag_f16(const _Float16* __restrict__ rowPtr,
                                                int k0, int laneHi) {
  const _Float16* p = rowPtr + k0 + (laneHi ? 16 : 0);
  v8h lo = *(const v8h*)(p);        // 16B
  v8h hi = *(const v8h*)(p + 8);    // 16B
  v16h a;
#pragma unroll
  for (int i = 0; i < 8; ++i) { a[i] = lo[i]; a[i + 8] = hi[i]; }
  return a;
}

// ---------------------------------------------------------------------------
// Weight prep: fp32 [K,N] -> f16 [N,K] (B-fragments become contiguous b128s)
// ---------------------------------------------------------------------------
__global__ __launch_bounds__(256) void cvt_f16_transpose(
    const float* __restrict__ src, _Float16* __restrict__ dst, int K, int N) {
  __shared__ float tile[32][33];
  const size_t plane = (size_t)K * N;
  const float* s = src + (size_t)blockIdx.z * plane;
  _Float16* d = dst + (size_t)blockIdx.z * plane;
  const int n0 = blockIdx.x * 32, k0 = blockIdx.y * 32;
  const int tx = threadIdx.x & 31, ty = threadIdx.x >> 5;  // 32x8
  for (int i = ty; i < 32; i += 8) {
    int k = k0 + i, n = n0 + tx;
    tile[i][tx] = (k < K && n < N) ? s[(size_t)k * N + n] : 0.0f;
  }
  __syncthreads();
  for (int i = ty; i < 32; i += 8) {
    int n = n0 + i, k = k0 + tx;
    if (n < N && k < K) d[(size_t)n * K + k] = (_Float16)tile[tx][i];
  }
}

__global__ __launch_bounds__(256) void cvt_f16_flat(
    const float* __restrict__ src, _Float16* __restrict__ dst, size_t n) {
  size_t i = (size_t)blockIdx.x * blockDim.x + threadIdx.x;
  size_t stride = (size_t)gridDim.x * blockDim.x;
  for (; i < n; i += stride) dst[i] = (_Float16)src[i];
}

// ---------------------------------------------------------------------------
// Embedding: x[b,s,:] = wte[id] + wpe[s]  (fp32 residual stream)
// ---------------------------------------------------------------------------
__global__ __launch_bounds__(256) void embed_kernel(
    const int* __restrict__ ids, const float* __restrict__ wte,
    const float* __restrict__ wpe, float* __restrict__ x) {
  const int bs = blockIdx.x;
  const int sPos = bs % S_;
  const int id = ids[bs];
  const float* we = wte + (size_t)id * D_;
  const float* pe = wpe + (size_t)sPos * D_;
  float* o = x + (size_t)bs * D_;
  for (int c = threadIdx.x; c < D_; c += 256) o[c] = we[c] + pe[c];
}

// ---------------------------------------------------------------------------
// LayerNorm: fp32 in, f16 out (LN output is only ever a WMMA A operand)
// ---------------------------------------------------------------------------
__global__ __launch_bounds__(256) void layernorm_kernel(
    const float* __restrict__ x, const float* __restrict__ g,
    const float* __restrict__ b, _Float16* __restrict__ out) {
  __shared__ float r1[256], r2[256];
  const int row = blockIdx.x, t = threadIdx.x;
  const float* p = x + (size_t)row * D_;
  float s = 0.f, s2 = 0.f;
  for (int c = t; c < D_; c += 256) { float v = p[c]; s += v; s2 += v * v; }
  r1[t] = s; r2[t] = s2;
  __syncthreads();
  for (int off = 128; off > 0; off >>= 1) {
    if (t < off) { r1[t] += r1[t + off]; r2[t] += r2[t + off]; }
    __syncthreads();
  }
  const float mean = r1[0] * (1.0f / D_);
  const float var = r2[0] * (1.0f / D_) - mean * mean;
  const float inv = rsqrtf(var + 1e-5f);
  _Float16* o = out + (size_t)row * D_;
  for (int c = t; c < D_; c += 256)
    o[c] = (_Float16)((p[c] - mean) * inv * g[c] + b[c]);
}

// ---------------------------------------------------------------------------
// WMMA GEMM: wave computes 16(M) x 64(N) via 4 accumulators sharing one A
// fragment. The K loop is unrolled x2 into a copy-free ping-pong: while the
// WMMA chain consumes buffer0 (block k), buffer1 (block k+32) is in flight,
// and buffer0 is refilled with block k+64 before buffer1 is consumed.
// Block = 8 waves (4 M-waves x 2 N-waves) -> 64x128 tile.
//   A f16 [M,K] row-major, Bt f16 [N,K] (pre-transposed), bias/resid fp32.
//   act: 1 = exact GELU.  Output: Ch (f16) if non-null else Cf (fp32).
//   vmode=1: scatter f16 output into [B, D, S] ("V transposed") layout.
// ---------------------------------------------------------------------------
__global__ __launch_bounds__(256) void gemm_wmma_kernel(
    const _Float16* __restrict__ A, const _Float16* __restrict__ Bt,
    const float* __restrict__ bias, const float* __restrict__ resid,
    float* __restrict__ Cf, _Float16* __restrict__ Ch,
    int M, int N, int K, int act, int vmode) {
  const int lane = threadIdx.x & 31;
  const int wave = threadIdx.x >> 5;
  const int laneHi = lane >> 4;
  const int lane16 = lane & 15;
  const int hiOff = laneHi ? 16 : 0;
  const int tileM = blockIdx.y * 64 + (wave >> 1) * 16;
  const int tileN0 = blockIdx.x * 128 + (wave & 1) * 64;

  const int rowA = tileM + lane16;
  const int aRow = (rowA < M) ? rowA : (M - 1);
  const _Float16* aPtr = A + (size_t)aRow * K;

  const _Float16* bPtr[4];
#pragma unroll
  for (int t = 0; t < 4; ++t) {
    const int colB = tileN0 + 16 * t + lane16;
    const int bCol = (colB < N) ? colB : (N - 1);
    bPtr[t] = Bt + (size_t)bCol * K;
  }

  v8f acc[4] = {};

  // ---- ping-pong K loop (x2 unroll, no inter-buffer register copies) ----
  v16h a0 = load_a_frag_f16(aPtr, 0, laneHi);
  v16h b0[4];
#pragma unroll
  for (int t = 0; t < 4; ++t) b0[t] = *(const v16h*)(bPtr[t] + hiOff);

  int k = 0;
  while (k + 32 < K) {
    if (k + 128 < K) {  // warm two pairs ahead (global_prefetch_b8)
      __builtin_prefetch((const void*)(aPtr + k + 128), 0, 0);
      __builtin_prefetch((const void*)(bPtr[0] + k + 128), 0, 0);
      __builtin_prefetch((const void*)(bPtr[1] + k + 128), 0, 0);
      __builtin_prefetch((const void*)(bPtr[2] + k + 128), 0, 0);
      __builtin_prefetch((const void*)(bPtr[3] + k + 128), 0, 0);
    }
    v16h a1 = load_a_frag_f16(aPtr, k + 32, laneHi);
    v16h b1[4];
#pragma unroll
    for (int t = 0; t < 4; ++t) b1[t] = *(const v16h*)(bPtr[t] + k + 32 + hiOff);
#pragma unroll
    for (int t = 0; t < 4; ++t)
      acc[t] = __builtin_amdgcn_wmma_f32_16x16x32_f16(false, a0, false, b0[t],
                                                      (short)0, acc[t], false, false);
    if (k + 64 < K) {
      a0 = load_a_frag_f16(aPtr, k + 64, laneHi);
#pragma unroll
      for (int t = 0; t < 4; ++t) b0[t] = *(const v16h*)(bPtr[t] + k + 64 + hiOff);
    }
#pragma unroll
    for (int t = 0; t < 4; ++t)
      acc[t] = __builtin_amdgcn_wmma_f32_16x16x32_f16(false, a1, false, b1[t],
                                                      (short)0, acc[t], false, false);
    k += 64;
  }
  if (k < K) {  // odd block count tail
#pragma unroll
    for (int t = 0; t < 4; ++t)
      acc[t] = __builtin_amdgcn_wmma_f32_16x16x32_f16(false, a0, false, b0[t],
                                                      (short)0, acc[t], false, false);
  }

#pragma unroll
  for (int t = 0; t < 4; ++t) {
#pragma unroll
    for (int r = 0; r < 8; ++r) {
      const int row = tileM + r + (laneHi ? 8 : 0);
      const int col = tileN0 + 16 * t + lane16;
      if (row < M && col < N) {
        float v = acc[t][r];
        if (bias) v += bias[col];
        if (act == 1) v = 0.5f * v * (1.0f + erff(v * 0.70710678118654752f));
        if (Ch) {
          if (vmode) {  // [B,S,D] -> [B, D, S] for contiguous attn-V B frags
            const int bb = row / S_, ss = row % S_;
            Ch[((size_t)(bb * D_ + col)) * S_ + ss] = (_Float16)v;
          } else {
            Ch[(size_t)row * N + col] = (_Float16)v;
          }
        } else {
          if (resid) v += resid[(size_t)row * N + col];
          Cf[(size_t)row * N + col] = v;
        }
      }
    }
  }
}

// ---------------------------------------------------------------------------
// Attention scores: scores[b,h,m,n] = (q_m . k_n)/8, causal-masked.
// q,k f16 [B,S,D]; wave tile 16x64; grid (S/128, S/64, B*H).
// K = HD = 64 -> 2 K-steps, fully unrolled by the compiler.
// ---------------------------------------------------------------------------
__global__ __launch_bounds__(256) void attn_scores_kernel(
    const _Float16* __restrict__ q, const _Float16* __restrict__ k,
    float* __restrict__ scores) {
  const int lane = threadIdx.x & 31;
  const int wave = threadIdx.x >> 5;
  const int laneHi = lane >> 4;
  const int lane16 = lane & 15;
  const int hiOff = laneHi ? 16 : 0;
  const int tileM = blockIdx.y * 64 + (wave >> 1) * 16;
  const int tileN0 = blockIdx.x * 128 + (wave & 1) * 64;
  const int bh = blockIdx.z;
  const int b = bh / H_, hh = bh % H_;

  const _Float16* qPtr =
      q + (size_t)(b * S_ + tileM + lane16) * D_ + hh * HD_;
  const _Float16* kPtr[4];
  bool live[4];
#pragma unroll
  for (int t = 0; t < 4; ++t) {
    kPtr[t] = k + (size_t)(b * S_ + tileN0 + 16 * t + lane16) * D_ + hh * HD_;
    live[t] = (tileN0 + 16 * t) <= (tileM + 15);
  }
  float* sPtr = scores + (size_t)bh * S_ * S_;

  v8f acc[4] = {};
  if (live[0]) {
#pragma unroll
    for (int k0 = 0; k0 < HD_; k0 += 32) {
      v16h a = load_a_frag_f16(qPtr, k0, laneHi);
#pragma unroll
      for (int t = 0; t < 4; ++t) {
        if (live[t]) {
          v16h bb = *(const v16h*)(kPtr[t] + k0 + hiOff);
          acc[t] = __builtin_amdgcn_wmma_f32_16x16x32_f16(
              false, a, false, bb, (short)0, acc[t], false, false);
        }
      }
    }
  }

  const float scale = 0.125f;  // 1/sqrt(64)
#pragma unroll
  for (int t = 0; t < 4; ++t) {
#pragma unroll
    for (int r = 0; r < 8; ++r) {
      const int row = tileM + r + (laneHi ? 8 : 0);
      const int col = tileN0 + 16 * t + lane16;
      float v = acc[t][r] * scale;
      if (col > row) v = NEGINF;
      sPtr[(size_t)row * S_ + col] = v;
    }
  }
}

// ---------------------------------------------------------------------------
// Row softmax: fp32 scores in, f16 probabilities out (WMMA A operand)
// ---------------------------------------------------------------------------
__global__ __launch_bounds__(256) void softmax_kernel(
    const float* __restrict__ s, _Float16* __restrict__ pr) {
  __shared__ float red[256];
  const int t = threadIdx.x;
  const float* p = s + (size_t)blockIdx.x * S_;
  _Float16* o = pr + (size_t)blockIdx.x * S_;
  float mx = -3.4e38f;
  for (int c = t; c < S_; c += 256) mx = fmaxf(mx, p[c]);
  red[t] = mx;
  __syncthreads();
  for (int off = 128; off > 0; off >>= 1) {
    if (t < off) red[t] = fmaxf(red[t], red[t + off]);
    __syncthreads();
  }
  mx = red[0];
  __syncthreads();
  float sum = 0.f;
  for (int c = t; c < S_; c += 256) sum += expf(p[c] - mx);
  red[t] = sum;
  __syncthreads();
  for (int off = 128; off > 0; off >>= 1) {
    if (t < off) red[t] += red[t + off];
    __syncthreads();
  }
  const float inv = 1.0f / red[0];
  for (int c = t; c < S_; c += 256) o[c] = (_Float16)(expf(p[c] - mx) * inv);
}

// ---------------------------------------------------------------------------
// o[b,m,h*64+n] = sum_k probs[b,h,m,k] * v[b,k,h*64+n]
// probs f16 [B*H,S,S]; vT f16 [B, D, S] (transposed by V projection epilogue)
// Block = 8 M-waves; wave tile 16(M) x 64(N=HD); grid (1, S/128, B*H).
// K loop truncated at the causal boundary; copy-free ping-pong (block count
// may be odd -> tail WMMA group).
// ---------------------------------------------------------------------------
__global__ __launch_bounds__(256) void attn_av_kernel(
    const _Float16* __restrict__ probs, const _Float16* __restrict__ vT,
    _Float16* __restrict__ o) {
  const int lane = threadIdx.x & 31;
  const int wave = threadIdx.x >> 5;
  const int laneHi = lane >> 4;
  const int lane16 = lane & 15;
  const int hiOff = laneHi ? 16 : 0;
  const int tileM = blockIdx.y * 128 + wave * 16;
  const int bh = blockIdx.z;
  const int b = bh / H_, hh = bh % H_;

  const _Float16* aPtr =
      probs + (size_t)bh * S_ * S_ + (size_t)(tileM + lane16) * S_;
  const _Float16* bPtr[4];
#pragma unroll
  for (int t = 0; t < 4; ++t)
    bPtr[t] = vT + (size_t)(b * D_ + hh * HD_ + 16 * t + lane16) * S_;

  // causal: probs row m is zero beyond col m  (kend >= 32 always)
  const int kend = (tileM + 16 + 31) & ~31;

  v8f acc[4] = {};
  v16h a0 = load_a_frag_f16(aPtr, 0, laneHi);
  v16h b0[4];
#pragma unroll
  for (int t = 0; t < 4; ++t) b0[t] = *(const v16h*)(bPtr[t] + hiOff);

  int k = 0;
  while (k + 32 < kend) {
    v16h a1 = load_a_frag_f16(aPtr, k + 32, laneHi);
    v16h b1[4];
#pragma unroll
    for (int t = 0; t < 4; ++t) b1[t] = *(const v16h*)(bPtr[t] + k + 32 + hiOff);
#pragma unroll
    for (int t = 0; t < 4; ++t)
      acc[t] = __builtin_amdgcn_wmma_f32_16x16x32_f16(false, a0, false, b0[t],
                                                      (short)0, acc[t], false, false);
    if (k + 64 < kend) {
      a0 = load_a_frag_f16(aPtr, k + 64, laneHi);
#pragma unroll
      for (int t = 0; t < 4; ++t) b0[t] = *(const v16h*)(bPtr[t] + k + 64 + hiOff);
    }
#pragma unroll
    for (int t = 0; t < 4; ++t)
      acc[t] = __builtin_amdgcn_wmma_f32_16x16x32_f16(false, a1, false, b1[t],
                                                      (short)0, acc[t], false, false);
    k += 64;
  }
  if (k < kend) {
#pragma unroll
    for (int t = 0; t < 4; ++t)
      acc[t] = __builtin_amdgcn_wmma_f32_16x16x32_f16(false, a0, false, b0[t],
                                                      (short)0, acc[t], false, false);
  }

#pragma unroll
  for (int t = 0; t < 4; ++t) {
#pragma unroll
    for (int r = 0; r < 8; ++r) {
      const int row = tileM + r + (laneHi ? 8 : 0);
      const int col = hh * HD_ + 16 * t + lane16;
      o[(size_t)(b * S_ + row) * D_ + col] = (_Float16)acc[t][r];
    }
  }
}

// ---------------------------------------------------------------------------
// Host orchestration
// ---------------------------------------------------------------------------
extern "C" void kernel_launch(void* const* d_in, const int* in_sizes, int n_in,
                              void* d_out, int out_size, void* d_ws, size_t ws_size,
                              hipStream_t stream) {
  (void)in_sizes; (void)n_in; (void)out_size; (void)ws_size;

  const int*   ids   = (const int*)d_in[0];
  const float* wte   = (const float*)d_in[1];
  const float* wpe   = (const float*)d_in[2];
  const float* ln1_g = (const float*)d_in[3];
  const float* ln1_b = (const float*)d_in[4];
  const float* Wq    = (const float*)d_in[5];
  const float* bq    = (const float*)d_in[6];
  const float* Wk    = (const float*)d_in[7];
  const float* bk    = (const float*)d_in[8];
  const float* Wv    = (const float*)d_in[9];
  const float* bv    = (const float*)d_in[10];
  const float* Wo    = (const float*)d_in[11];
  const float* bo    = (const float*)d_in[12];
  const float* ln2_g = (const float*)d_in[13];
  const float* ln2_b = (const float*)d_in[14];
  const float* W1    = (const float*)d_in[15];
  const float* b1    = (const float*)d_in[16];
  const float* W2    = (const float*)d_in[17];
  const float* b2    = (const float*)d_in[18];
  const float* lnf_g = (const float*)d_in[19];
  const float* lnf_b = (const float*)d_in[20];

  char* ws = (char*)d_ws;
  size_t off = 0;
  auto alloc = [&](size_t bytes) -> void* {
    off = (off + 255) & ~(size_t)255;
    void* p = ws + off;
    off += bytes;
    return p;
  };

  _Float16* wte16 = (_Float16*)alloc((size_t)V_ * D_ * sizeof(_Float16));
  _Float16* wq16  = (_Float16*)alloc((size_t)L_ * D_ * D_ * sizeof(_Float16));
  _Float16* wk16  = (_Float16*)alloc((size_t)L_ * D_ * D_ * sizeof(_Float16));
  _Float16* wv16  = (_Float16*)alloc((size_t)L_ * D_ * D_ * sizeof(_Float16));
  _Float16* wo16  = (_Float16*)alloc((size_t)L_ * D_ * D_ * sizeof(_Float16));
  _Float16* w116  = (_Float16*)alloc((size_t)L_ * D_ * FF_ * sizeof(_Float16));
  _Float16* w216  = (_Float16*)alloc((size_t)L_ * D_ * FF_ * sizeof(_Float16));
  float*    xbuf  = (float*)alloc((size_t)M_ * D_ * sizeof(float));
  _Float16* hbuf  = (_Float16*)alloc((size_t)M_ * D_ * sizeof(_Float16));
  _Float16* qbuf  = (_Float16*)alloc((size_t)M_ * D_ * sizeof(_Float16));
  _Float16* kbuf  = (_Float16*)alloc((size_t)M_ * D_ * sizeof(_Float16));
  _Float16* vT    = (_Float16*)alloc((size_t)B_ * D_ * S_ * sizeof(_Float16));
  _Float16* obuf  = (_Float16*)alloc((size_t)M_ * D_ * sizeof(_Float16));
  _Float16* ffbuf = (_Float16*)alloc((size_t)M_ * FF_ * sizeof(_Float16));
  float*    sc    = (float*)alloc((size_t)B_ * H_ * S_ * S_ * sizeof(float));
  _Float16* scp   = (_Float16*)alloc((size_t)B_ * H_ * S_ * S_ * sizeof(_Float16));

  const dim3 blk(256);

  // ---- weight prep: f16, transposed to [N,K] ----
  cvt_f16_flat<<<4096, blk, 0, stream>>>(wte, wte16, (size_t)V_ * D_);
  {
    dim3 gDD(D_ / 32, D_ / 32, L_);
    cvt_f16_transpose<<<gDD, blk, 0, stream>>>(Wq, wq16, D_, D_);
    cvt_f16_transpose<<<gDD, blk, 0, stream>>>(Wk, wk16, D_, D_);
    cvt_f16_transpose<<<gDD, blk, 0, stream>>>(Wv, wv16, D_, D_);
    cvt_f16_transpose<<<gDD, blk, 0, stream>>>(Wo, wo16, D_, D_);
    dim3 g1(FF_ / 32, D_ / 32, L_);   // W1 [D,FF] -> [FF,D]
    cvt_f16_transpose<<<g1, blk, 0, stream>>>(W1, w116, D_, FF_);
    dim3 g2(D_ / 32, FF_ / 32, L_);   // W2 [FF,D] -> [D,FF]
    cvt_f16_transpose<<<g2, blk, 0, stream>>>(W2, w216, FF_, D_);
  }

  embed_kernel<<<M_, blk, 0, stream>>>(ids, wte, wpe, xbuf);

  const dim3 gProj((D_ + 127) / 128, M_ / 64);   // N=768
  const dim3 gFF1((FF_ + 127) / 128, M_ / 64);   // N=3072
  const dim3 gScores(S_ / 128, S_ / 64, B_ * H_);
  const dim3 gAV(1, S_ / 128, B_ * H_);

  for (int l = 0; l < L_; ++l) {
    const _Float16* wq = wq16 + (size_t)l * D_ * D_;
    const _Float16* wk = wk16 + (size_t)l * D_ * D_;
    const _Float16* wv = wv16 + (size_t)l * D_ * D_;
    const _Float16* wo = wo16 + (size_t)l * D_ * D_;
    const _Float16* w1 = w116 + (size_t)l * D_ * FF_;
    const _Float16* w2 = w216 + (size_t)l * D_ * FF_;

    layernorm_kernel<<<M_, blk, 0, stream>>>(xbuf, ln1_g + (size_t)l * D_,
                                             ln1_b + (size_t)l * D_, hbuf);
    gemm_wmma_kernel<<<gProj, blk, 0, stream>>>(hbuf, wq, bq + (size_t)l * D_,
                                                nullptr, nullptr, qbuf,
                                                M_, D_, D_, 0, 0);
    gemm_wmma_kernel<<<gProj, blk, 0, stream>>>(hbuf, wk, bk + (size_t)l * D_,
                                                nullptr, nullptr, kbuf,
                                                M_, D_, D_, 0, 0);
    gemm_wmma_kernel<<<gProj, blk, 0, stream>>>(hbuf, wv, bv + (size_t)l * D_,
                                                nullptr, nullptr, vT,
                                                M_, D_, D_, 0, 1);  // transposed

    attn_scores_kernel<<<gScores, blk, 0, stream>>>(qbuf, kbuf, sc);
    softmax_kernel<<<B_ * H_ * S_, blk, 0, stream>>>(sc, scp);
    attn_av_kernel<<<gAV, blk, 0, stream>>>(scp, vT, obuf);

    gemm_wmma_kernel<<<gProj, blk, 0, stream>>>(obuf, wo, bo + (size_t)l * D_,
                                                xbuf, xbuf, nullptr,
                                                M_, D_, D_, 0, 0);

    layernorm_kernel<<<M_, blk, 0, stream>>>(xbuf, ln2_g + (size_t)l * D_,
                                             ln2_b + (size_t)l * D_, hbuf);
    gemm_wmma_kernel<<<gFF1, blk, 0, stream>>>(hbuf, w1, b1 + (size_t)l * FF_,
                                               nullptr, nullptr, ffbuf,
                                               M_, FF_, D_, 1, 0);
    gemm_wmma_kernel<<<gProj, blk, 0, stream>>>(ffbuf, w2, b2 + (size_t)l * D_,
                                                xbuf, xbuf, nullptr,
                                                M_, D_, FF_, 0, 0);
  }

  layernorm_kernel<<<M_, blk, 0, stream>>>(xbuf, lnf_g, lnf_b, hbuf);
  const dim3 gHead((V_ + 127) / 128, M_ / 64);
  gemm_wmma_kernel<<<gHead, blk, 0, stream>>>(hbuf, wte16, nullptr, nullptr,
                                              (float*)d_out, nullptr,
                                              M_, V_, D_, 0, 0);
}
